// DistMap_63256278335589
// MI455X (gfx1250) — matrix operand order repaired and checked by reading, not compiled, not run on hardware
//
#include <hip/hip_runtime.h>

// DistMap: d[b,i,j] = |a_i|^2 + |b_j|^2 - 2<a_i,b_j>
//   a,b: [4, 64, 4096] f32;  out: [4, 4096, 4096] f32
// Strategy: split-bf16 GEMM (hi/lo decomposition, 3 WMMA products) on
// v_wmma_f32_16x16x32_bf16, inputs repacked into fragment-native layout,
// norms fused in epilogue.

typedef __attribute__((ext_vector_type(16))) __bf16 v16bf;
typedef __attribute__((ext_vector_type(8)))  float  v8f;

#define B_  4
#define C_  64
#define N_  4096
#define M_  4096
// packed fragment arrays: element u16 index layout
//   (((b*2 + kb)*256 + tile16)*32 + lane)*16 + slot   -> total B_*C_*N_ = 1M u16
#define PACK_BYTES (B_ * C_ * N_ * 2)   // 2 MB per array

__device__ __forceinline__ unsigned bf16_rne(float x) {
    unsigned u = __float_as_uint(x);
    return (u + 0x7FFFu + ((u >> 16) & 1u)) >> 16;   // round-to-nearest-even
}

// ---------------------------------------------------------------------------
// Prep: split f32 -> (hi,lo) bf16 and scatter into WMMA-fragment-native layout.
// A-operand lane/K map (16-bit A 16x32):  lanes 0-15: K={s<8 ? s : s+8}
//                                         lanes 16-31: K=8+{...}
// B-operand lane/K map (16-bit B 32x16):  same as A
// ---------------------------------------------------------------------------
__global__ __launch_bounds__(256) void distmap_pack(
    const float* __restrict__ a, const float* __restrict__ bsrc,
    unsigned short* __restrict__ aHi, unsigned short* __restrict__ aLo,
    unsigned short* __restrict__ bHi, unsigned short* __restrict__ bLo) {
    unsigned t = blockIdx.x * blockDim.x + threadIdx.x;
    const unsigned per = B_ * C_ * N_;            // 1,048,576
    bool isB = t >= per;
    unsigned u = isB ? (t - per) : t;

    unsigned s    = u & 15u;           // bf16 slot within lane (0..15)
    unsigned lane = (u >> 4) & 31u;    // lane (0..31)
    unsigned nt   = (u >> 9) & 255u;   // 16-wide tile along N/M
    unsigned kb   = (u >> 17) & 1u;    // K block of 32 (C=64 -> 2 blocks)
    unsigned b    = u >> 18;           // batch

    unsigned col = nt * 16u + (lane & 15u);       // n (A-side) or m (B-side)
    unsigned kLocal;
    if (!isB) kLocal = ((lane < 16u) ? 0u : 8u)  + ((s < 8u) ? s : s + 8u);
    else      kLocal = ((lane < 16u) ? 0u : 8u)  + ((s < 8u) ? s : s + 8u);
    unsigned c = kb * 32u + kLocal;

    const float* src = isB ? bsrc : a;
    float x = src[(b * C_ + c) * N_ + col];

    unsigned hiBits = bf16_rne(x);
    float    hif    = __uint_as_float(hiBits << 16);
    unsigned loBits = bf16_rne(x - hif);

    if (!isB) { aHi[u] = (unsigned short)hiBits; aLo[u] = (unsigned short)loBits; }
    else      { bHi[u] = (unsigned short)hiBits; bLo[u] = (unsigned short)loBits; }
}

// ---------------------------------------------------------------------------
// Norms: aa[b,n] = sum_c a[b,c,n]^2 ; bb[b,m] = sum_c b[b,c,m]^2 (exact f32)
// ---------------------------------------------------------------------------
__global__ __launch_bounds__(256) void distmap_norms(
    const float* __restrict__ a, const float* __restrict__ bsrc,
    float* __restrict__ aa, float* __restrict__ bb) {
    unsigned t = blockIdx.x * blockDim.x + threadIdx.x;   // 0..2*B_*N_-1
    bool isB = t >= (unsigned)(B_ * N_);
    unsigned u = isB ? (t - B_ * N_) : t;
    unsigned n = u & (N_ - 1);
    unsigned b = u >> 12;                                  // N_ = 4096 = 2^12
    const float* src = isB ? bsrc : a;
    float s = 0.f;
    #pragma unroll 8
    for (int c = 0; c < C_; ++c) {
        float x = src[(b * C_ + c) * N_ + n];
        s = fmaf(x, x, s);
    }
    (isB ? bb : aa)[u] = s;
}

// ---------------------------------------------------------------------------
// GEMM + epilogue. 256 threads = 8 wave32s. WG tile 128x128, wave tile 32x64.
// 48 v_wmma_f32_16x16x32_bf16 per wave (2 ksteps x 8 C-tiles x 3 split terms).
// ---------------------------------------------------------------------------
__global__ __launch_bounds__(256) void distmap_gemm(
    const v16bf* __restrict__ aHi, const v16bf* __restrict__ aLo,
    const v16bf* __restrict__ bHi, const v16bf* __restrict__ bLo,
    const float* __restrict__ aa, const float* __restrict__ bb,
    float* __restrict__ out) {
    const int lane = threadIdx.x & 31;
    const int wave = threadIdx.x >> 5;
    const int wn   = wave & 3;          // 4 waves along N
    const int wm   = wave >> 2;         // 2 waves along M
    const int bIdx = blockIdx.z;
    const int bn   = blockIdx.x;        // N/128
    const int bm   = blockIdx.y;        // M/128

    v8f acc[2][4];
    #pragma unroll
    for (int i = 0; i < 2; ++i)
        #pragma unroll
        for (int j = 0; j < 4; ++j) acc[i][j] = (v8f)0.0f;

    #pragma unroll
    for (int kb = 0; kb < 2; ++kb) {
        const int kbase = (bIdx * 2 + kb) * 256;
        v16bf aH[2], aL[2], bH[4], bL[4];
        #pragma unroll
        for (int i = 0; i < 2; ++i) {
            size_t idx = (size_t)(kbase + bn * 8 + wn * 2 + i) * 32 + lane;
            aH[i] = aHi[idx];  aL[i] = aLo[idx];   // 2x global_load_b128 each
        }
        #pragma unroll
        for (int j = 0; j < 4; ++j) {
            size_t idx = (size_t)(kbase + bm * 8 + wm * 4 + j) * 32 + lane;
            bH[j] = bHi[idx];  bL[j] = bLo[idx];
        }
        // p=0: hi*hi   p=1: hi*lo   p=2: lo*hi  (lo*lo negligible)
        #pragma unroll
        for (int p = 0; p < 3; ++p) {
            #pragma unroll
            for (int i = 0; i < 2; ++i) {
                #pragma unroll
                for (int j = 0; j < 4; ++j) {
                    const v16bf& A  = (p == 2) ? aL[i] : aH[i];
                    const v16bf& Bm = (p == 1) ? bL[j] : bH[j];
                    acc[i][j] = __builtin_amdgcn_wmma_f32_16x16x32_bf16(
                        false, A, false, Bm, (short)0, acc[i][j], false, false);
                }
            }
        }
    }

    // Epilogue: d = aa + bb - 2*ab.
    // C/D layout: VGPR r -> row (lane<16 ? r : r+8), col = lane%16.
    const int halfSel = (lane < 16) ? 0 : 8;
    #pragma unroll
    for (int i = 0; i < 2; ++i) {
        const int nBase = bn * 128 + wn * 32 + i * 16 + halfSel;
        float aaV[8];
        #pragma unroll
        for (int r = 0; r < 8; ++r) aaV[r] = aa[bIdx * N_ + nBase + r];
        #pragma unroll
        for (int j = 0; j < 4; ++j) {
            const int m  = bm * 128 + wm * 64 + j * 16 + (lane & 15);
            const float bbV = bb[bIdx * M_ + m];
            #pragma unroll
            for (int r = 0; r < 8; ++r) {
                const int n = nBase + r;
                out[((size_t)bIdx * N_ + n) * M_ + m] =
                    aaV[r] + bbV - 2.0f * acc[i][j][r];
            }
        }
    }
}

extern "C" void kernel_launch(void* const* d_in, const int* in_sizes, int n_in,
                              void* d_out, int out_size, void* d_ws, size_t ws_size,
                              hipStream_t stream) {
    const float* a = (const float*)d_in[0];
    const float* b = (const float*)d_in[1];
    float* out = (float*)d_out;

    // Workspace map (needs 8.125 MB):
    //   [0,2M)   aHi   [2M,4M) aLo   [4M,6M) bHi   [6M,8M) bLo
    //   [8M, +64K) aa   [+64K, +128K) bb
    char* ws = (char*)d_ws;
    unsigned short* aHi = (unsigned short*)(ws + 0 * (size_t)PACK_BYTES);
    unsigned short* aLo = (unsigned short*)(ws + 1 * (size_t)PACK_BYTES);
    unsigned short* bHi = (unsigned short*)(ws + 2 * (size_t)PACK_BYTES);
    unsigned short* bLo = (unsigned short*)(ws + 3 * (size_t)PACK_BYTES);
    float* aa = (float*)(ws + 4 * (size_t)PACK_BYTES);
    float* bb = (float*)(ws + 4 * (size_t)PACK_BYTES + B_ * N_ * sizeof(float));

    // 1) split + repack into fragment-native bf16 (2M threads)
    distmap_pack<<<(2 * B_ * C_ * N_) / 256, 256, 0, stream>>>(a, b, aHi, aLo, bHi, bLo);
    // 2) exact f32 squared norms (32768 threads)
    distmap_norms<<<(2 * B_ * N_) / 256, 256, 0, stream>>>(a, b, aa, bb);
    // 3) WMMA GEMM + fused epilogue
    dim3 grid(N_ / 128, M_ / 128, B_);
    distmap_gemm<<<grid, 256, 0, stream>>>((const v16bf*)aHi, (const v16bf*)aLo,
                                           (const v16bf*)bHi, (const v16bf*)bLo,
                                           aa, bb, out);
}
